// MACE_21139829031606
// MI455X (gfx1250) — matrix operand order repaired and checked by reading, not compiled, not run on hardware
//
#include <hip/hip_runtime.h>
#include <math.h>
#include <type_traits>

// ---------------------------------------------------------------------------
// MACE-like GNN forward for MI455X (gfx1250).
//   * [M,64]@[64,64] GEMMs on v_wmma_f32_16x16x32_f16 (f32 accumulate),
//     weights pre-packed into WMMA B-fragment order -> 32B vector loads,
//     epilogue uses tile-uniform guard + immediate-offset stores.
//   * Edge scatter: global_atomic_add_f32 (dominant, ~0.8 GB -> HBM-bound).
// ---------------------------------------------------------------------------

#define C64 64
#define NB 8
#define LM 9
#define R_MAX 5.0f
#define PI_F 3.14159265358979f

typedef __attribute__((ext_vector_type(16))) _Float16 v16h;  // 32 B
typedef __attribute__((ext_vector_type(8)))  _Float16 v8h;   // 16 B
typedef __attribute__((ext_vector_type(8)))  float    v8f;   // 32 B

// -------------------------------- small utility kernels --------------------

__global__ void zero_f32_kernel(float* __restrict__ p, int n) {
    int i = blockIdx.x * blockDim.x + threadIdx.x;
    if (i < n) p[i] = 0.0f;
}

// Convert f32 weight matrices [nmat][64][64] into f16 WMMA B-fragment order:
//   P[mat*4096 + ((nt*2+kc)*32 + lane)*16 + j] =
//       B[mat*4096 + (kc*32 + (j>>3)*16 + (lane>>4)*8 + (j&7))*64 + nt*16 + (lane&15)]
__global__ void pack_b_kernel(const float* __restrict__ B, _Float16* __restrict__ P, int nmat) {
    int gid = blockIdx.x * blockDim.x + threadIdx.x;
    if (gid >= nmat * 4096) return;
    int mat  = gid >> 12;
    int idx  = gid & 4095;
    int j    = idx & 15;
    int lane = (idx >> 4) & 31;
    int frag = idx >> 9;               // nt*2 + kc
    int nt = frag >> 1, kc = frag & 1;
    int k = kc * 32 + (j >> 3) * 16 + (lane >> 4) * 8 + (j & 7);
    int n = nt * 16 + (lane & 15);
    P[gid] = (_Float16)B[mat * 4096 + k * C64 + n];
}

// h[i,c] = W_embed[atom_types[i], c]
__global__ void embed_kernel(const float* __restrict__ W_embed, const int* __restrict__ types,
                             float* __restrict__ h, int Nn) {
    int gid = blockIdx.x * blockDim.x + threadIdx.x;
    if (gid >= Nn * C64) return;
    int i = gid >> 6, c = gid & 63;
    h[gid] = W_embed[types[i] * C64 + c];
}

// -------------------------------- edge geometry + radial MLP layer 1 -------

__global__ void edge_geom_kernel(const float* __restrict__ pos, const int* __restrict__ ei,
                                 const float* __restrict__ W_r1, const float* __restrict__ b_r1,
                                 float* __restrict__ sh, _Float16* __restrict__ hidden, int En) {
    int e = blockIdx.x * blockDim.x + threadIdx.x;
    if (e >= En) return;
    int snd = ei[e];
    int rcv = ei[En + e];
    float dx = pos[rcv * 3 + 0] - pos[snd * 3 + 0];
    float dy = pos[rcv * 3 + 1] - pos[snd * 3 + 1];
    float dz = pos[rcv * 3 + 2] - pos[snd * 3 + 2];
    float r  = sqrtf(dx * dx + dy * dy + dz * dz);
    float in = 1.0f / fmaxf(r, 1e-9f);
    float x = dx * in, y = dy * in, z = dz * in;

    const float s3 = 1.7320508f;
    float shv[LM];
    shv[0] = 1.0f;  shv[1] = x;  shv[2] = y;  shv[3] = z;
    shv[4] = s3 * x * y;  shv[5] = s3 * y * z;
    shv[6] = 0.5f * (3.0f * z * z - 1.0f);
    shv[7] = s3 * x * z;
    shv[8] = 0.5f * s3 * (x * x - y * y);
#pragma unroll
    for (int j = 0; j < LM; ++j) sh[e * LM + j] = shv[j];

    // radial bessel embedding with polynomial cutoff envelope
    float rs   = fmaxf(r, 1e-9f);
    float pref = sqrtf(2.0f / R_MAX);
    float xx   = r / R_MAX;
    float env  = 0.0f;
    if (xx < 1.0f) {
        float x2 = xx * xx, x6 = x2 * x2 * x2, x7 = x6 * xx, x8 = x7 * xx;
        env = 1.0f - 28.0f * x6 + 48.0f * x7 - 21.0f * x8;
    }
    // sin((n+1)t) via Chebyshev recurrence: one sincos instead of 8 sinf
    float t = PI_F * r / R_MAX;
    float s1, c1;
    __sincosf(t, &s1, &c1);
    float scale = pref * env / rs;
    float ef[NB];
    float sn = s1, snm1 = 0.0f;
#pragma unroll
    for (int n = 0; n < NB; ++n) {
        ef[n] = scale * sn;
        float snext = 2.0f * c1 * sn - snm1;
        snm1 = sn;
        sn = snext;
    }

    // hidden = silu(ef @ W_r1 + b_r1), vectorized 16B stores
    _Float16* hrow = hidden + e * C64;
#pragma unroll
    for (int j0 = 0; j0 < C64; j0 += 8) {
        v8h o;
#pragma unroll
        for (int jj = 0; jj < 8; ++jj) {
            int j = j0 + jj;
            float a = b_r1[j];
#pragma unroll
            for (int n = 0; n < NB; ++n) a += ef[n] * W_r1[n * C64 + j];
            o[jj] = (_Float16)(a / (1.0f + expf(-a)));
        }
        *(v8h*)(hrow + j0) = o;
    }
}

// -------------------------------- WMMA GEMM:  D[M,64] = A[M,64] @ Bpacked (+Src)

template <typename AT, typename OT, bool ADD>
__global__ __launch_bounds__(128)
void wmma_gemm64_kernel(const AT* __restrict__ Am, const _Float16* __restrict__ Bp,
                        const float* __restrict__ Src, OT* __restrict__ Dm, int M) {
    const int wavesPerBlock = blockDim.x >> 5;
    const int wave0   = blockIdx.x * wavesPerBlock + ((int)threadIdx.x >> 5);
    const int nWaves  = gridDim.x * wavesPerBlock;
    const int nTiles  = (M + 15) >> 4;
    const int lane = threadIdx.x & 31;
    const int half = lane >> 4;
    const int l16  = lane & 15;

    // ---- preload all 8 B fragments: one aligned 32B vector load each
    const v16h* Bv = (const v16h*)Bp;
    v16h bfr[4][2];
#pragma unroll
    for (int nt = 0; nt < 4; ++nt)
#pragma unroll
        for (int kc = 0; kc < 2; ++kc)
            bfr[nt][kc] = Bv[(nt * 2 + kc) * 32 + lane];

    for (int tile = wave0; tile < nTiles; tile += nWaves) {
        int row0 = tile << 4;
        int arow = row0 + l16;
        if (arow >= M) arow = M - 1;

        // ---- A fragments: two contiguous granules per K-chunk
        v16h afr[2];
        if constexpr (std::is_same<AT, _Float16>::value) {
            const v8h* Ar = (const v8h*)(Am + arow * C64);
#pragma unroll
            for (int kc = 0; kc < 2; ++kc) {
                v8h lo = Ar[kc * 4 + half];
                v8h hi = Ar[kc * 4 + half + 2];
                afr[kc] = __builtin_shufflevector(lo, hi, 0, 1, 2, 3, 4, 5, 6, 7,
                                                  8, 9, 10, 11, 12, 13, 14, 15);
            }
        } else {
            const v8f* Ar = (const v8f*)(Am + arow * C64);
#pragma unroll
            for (int kc = 0; kc < 2; ++kc) {
                v8f lo = Ar[kc * 4 + half];
                v8f hi = Ar[kc * 4 + half + 2];
#pragma unroll
                for (int j = 0; j < 8; ++j) {
                    afr[kc][j]     = (_Float16)lo[j];
                    afr[kc][8 + j] = (_Float16)hi[j];
                }
            }
        }

        // ---- 4 N-tiles x 2 K-chunks = 8 WMMAs; accumulate all before epilogue
        v8f acc4[4];
#pragma unroll
        for (int nt = 0; nt < 4; ++nt) {
            v8f acc = {};
            acc = __builtin_amdgcn_wmma_f32_16x16x32_f16(false, afr[0], false, bfr[nt][0],
                                                         (short)0, acc, false, false);
            acc = __builtin_amdgcn_wmma_f32_16x16x32_f16(false, afr[1], false, bfr[nt][1],
                                                         (short)0, acc, false, false);
            acc4[nt] = acc;
        }

        // ---- epilogue: lane base = (row0 + half*8)*64 + l16, then immediate
        // offsets v*64 + nt*16 (folded into the store encoding after unroll).
        // D layout: VGPR v -> M = row0 + half*8 + v ; N = nt*16 + l16.
        int base = (row0 + half * 8) * C64 + l16;
        if (row0 + 16 <= M) {                       // full tile: unguarded
            OT* dbase = Dm + base;
            const float* sbase = ADD ? (Src + base) : nullptr;
#pragma unroll
            for (int nt = 0; nt < 4; ++nt)
#pragma unroll
                for (int v = 0; v < 8; ++v) {
                    float val = acc4[nt][v];
                    if (ADD) val += sbase[v * C64 + nt * 16];
                    dbase[v * C64 + nt * 16] = (OT)val;
                }
        } else {                                    // ragged tail (unused here)
#pragma unroll
            for (int nt = 0; nt < 4; ++nt)
#pragma unroll
                for (int v = 0; v < 8; ++v) {
                    int row = row0 + half * 8 + v;
                    if (row < M) {
                        float val = acc4[nt][v];
                        if (ADD) val += Src[row * C64 + nt * 16 + l16];
                        Dm[row * C64 + nt * 16 + l16] = (OT)val;
                    }
                }
        }
    }
}

// -------------------------------- message scatter (dominant, atomic-bound) --

__global__ void message_kernel(const _Float16* __restrict__ R, const float* __restrict__ h1,
                               const float* __restrict__ sh, const int* __restrict__ ei,
                               float* __restrict__ A, int En) {
    int gid = blockIdx.x * blockDim.x + threadIdx.x;
    if (gid >= En * C64) return;
    int e = gid >> 6, c = gid & 63;
    int snd = ei[e];
    int rcv = ei[En + e];
    float t = (float)R[e * C64 + c] * h1[snd * C64 + c] * (1.0f / 16.0f);
    const float* she = sh + e * LM;
    float* Ar = A + rcv * (LM * C64) + c;
#pragma unroll
    for (int lm = 0; lm < LM; ++lm)
        atomicAdd(Ar + lm * C64, she[lm] * t);
}

// q[i,c] = A[i,0,c] + sum_lm A[i,lm,c]^2
__global__ void q_kernel(const float* __restrict__ A, float* __restrict__ q, int Nn) {
    int gid = blockIdx.x * blockDim.x + threadIdx.x;
    if (gid >= Nn * C64) return;
    int i = gid >> 6, c = gid & 63;
    const float* Ai = A + i * (LM * C64) + c;
    float a0 = Ai[0];
    float s  = 0.0f;
#pragma unroll
    for (int lm = 0; lm < LM; ++lm) {
        float v = Ai[lm * C64];
        s += v * v;
    }
    q[gid] = a0 + s;
}

// out[batch[i]] += h[i,:] . w_ro
__global__ void readout_kernel(const float* __restrict__ h, const float* __restrict__ wro,
                               const int* __restrict__ batch, float* __restrict__ out, int Nn) {
    int i = blockIdx.x * blockDim.x + threadIdx.x;
    if (i >= Nn) return;
    const float4* h4 = (const float4*)(h + i * C64);
    const float4* w4 = (const float4*)wro;
    float e = 0.0f;
#pragma unroll
    for (int c = 0; c < C64 / 4; ++c) {
        float4 a = h4[c], b = w4[c];
        e += a.x * b.x + a.y * b.y + a.z * b.z + a.w * b.w;
    }
    atomicAdd(out + batch[i], e);
}

// -------------------------------- host orchestration ------------------------

extern "C" void kernel_launch(void* const* d_in, const int* in_sizes, int n_in,
                              void* d_out, int out_size, void* d_ws, size_t ws_size,
                              hipStream_t stream) {
    const float* pos     = (const float*)d_in[0];
    const float* W_embed = (const float*)d_in[1];
    const float* W_r1    = (const float*)d_in[2];
    const float* b_r1    = (const float*)d_in[3];
    const float* W_r2    = (const float*)d_in[4];
    const float* W_up    = (const float*)d_in[5];
    const float* W_sc    = (const float*)d_in[6];
    const float* W_prod  = (const float*)d_in[7];
    const float* w_ro    = (const float*)d_in[8];
    const int*   types   = (const int*)d_in[9];
    const int*   ei      = (const int*)d_in[10];
    const int*   batch   = (const int*)d_in[11];

    const int N  = in_sizes[9];
    const int E  = in_sizes[10] / 2;
    const int Lh = 2;
    float* out = (float*)d_out;

    // ---- workspace carve-up (bytes, 256-aligned)
    size_t off = 0;
    auto carve = [&](size_t bytes) -> void* {
        void* p = (char*)d_ws + off;
        off = (off + bytes + 255) & ~(size_t)255;
        return p;
    };
    _Float16* Wr2h   = (_Float16*)carve(4096 * 2);          // packed B fragments
    _Float16* Wuph   = (_Float16*)carve(2 * 4096 * 2);
    _Float16* Wsch   = (_Float16*)carve(2 * 4096 * 2);
    _Float16* Wprodh = (_Float16*)carve(2 * 4096 * 2);
    float* h   = (float*)carve((size_t)N * C64 * 4);
    float* h1  = (float*)carve((size_t)N * C64 * 4);
    float* sc  = (float*)carve((size_t)N * C64 * 4);
    float* q   = (float*)carve((size_t)N * C64 * 4);
    float* sh  = (float*)carve((size_t)E * LM * 4);
    _Float16* Rh = (_Float16*)carve((size_t)E * C64 * 2);
    float* Abuf  = (float*)carve((size_t)N * LM * C64 * 4);
    // `hidden` (E*64 f16 = 20.5MB) is dead before A is zeroed -> alias A (23MB)
    _Float16* hidden = (_Float16*)Abuf;

    const int TPB = 256;
    auto blocks = [](long n, int tpb) { return (int)((n + tpb - 1) / tpb); };

    // weight conversion + fragment packing (f32 -> f16, WMMA B layout)
    pack_b_kernel<<<blocks(4096, TPB), TPB, 0, stream>>>(W_r2, Wr2h, 1);
    pack_b_kernel<<<blocks(2 * 4096, TPB), TPB, 0, stream>>>(W_up, Wuph, 2);
    pack_b_kernel<<<blocks(2 * 4096, TPB), TPB, 0, stream>>>(W_sc, Wsch, 2);
    pack_b_kernel<<<blocks(2 * 4096, TPB), TPB, 0, stream>>>(W_prod, Wprodh, 2);

    // node embedding + edge geometry / radial MLP layer 1
    embed_kernel<<<blocks((long)N * C64, TPB), TPB, 0, stream>>>(W_embed, types, h, N);
    edge_geom_kernel<<<blocks(E, TPB), TPB, 0, stream>>>(pos, ei, W_r1, b_r1, sh, hidden, E);

    // R = hidden @ W_r2  (f16 in, f16 out, WMMA, persistent waves)
    {
        int tiles = (E + 15) >> 4;
        int blks  = (tiles + 3) / 4;
        if (blks > 640) blks = 640;
        wmma_gemm64_kernel<_Float16, _Float16, false>
            <<<blks, 128, 0, stream>>>(hidden, Wr2h, nullptr, Rh, E);
    }

    // zero output accumulator (pair_energy term is identically zero)
    zero_f32_kernel<<<blocks(out_size, TPB), TPB, 0, stream>>>(out, out_size);

    int ntiles = (N + 15) >> 4;
    int ngemm  = (ntiles + 3) / 4;
    if (ngemm > 640) ngemm = 640;
    for (int l = 0; l < Lh; ++l) {
        // h1 = h @ W_up[l] ; sc = h @ W_sc[l]   (WMMA)
        wmma_gemm64_kernel<float, float, false>
            <<<ngemm, 128, 0, stream>>>(h, Wuph + l * 4096, nullptr, h1, N);
        wmma_gemm64_kernel<float, float, false>
            <<<ngemm, 128, 0, stream>>>(h, Wsch + l * 4096, nullptr, sc, N);

        // A = segment_sum(sh * (R * h1[sender]), receiver) / 16
        zero_f32_kernel<<<blocks((long)N * LM * C64, TPB), TPB, 0, stream>>>(Abuf, N * LM * C64);
        message_kernel<<<blocks((long)E * C64, TPB), TPB, 0, stream>>>(Rh, h1, sh, ei, Abuf, E);

        // q = A[:,0,:] + sum_lm A^2 ;  h = q @ W_prod[l] + sc   (WMMA fused add)
        q_kernel<<<blocks((long)N * C64, TPB), TPB, 0, stream>>>(Abuf, q, N);
        wmma_gemm64_kernel<float, float, true>
            <<<ngemm, 128, 0, stream>>>(q, Wprodh + l * 4096, sc, h, N);

        // out[batch[i]] += h[i,:] . w_ro[l]
        readout_kernel<<<blocks(N, TPB), TPB, 0, stream>>>(h, w_ro + l * C64, batch, out, N);
    }
}
